// rel_aux_net_13941463842948
// MI455X (gfx1250) — matrix operand order repaired and checked by reading, not compiled, not run on hardware
//
#include <hip/hip_runtime.h>
#include <hip/hip_bf16.h>
#include <math.h>

typedef _Float16 v16h __attribute__((ext_vector_type(16)));
typedef float    v8f  __attribute__((ext_vector_type(8)));
typedef unsigned int v4u __attribute__((ext_vector_type(4)));

union Frag16 { v16h h; v4u q[2]; };

__device__ __forceinline__ v16h ld_frag(const void* p0, const void* p1) {
  Frag16 f;
  f.q[0] = *(const v4u*)p0;
  f.q[1] = *(const v4u*)p1;
  return f.h;
}

// Problem constants (fixed by the reference)
constexpr int kB   = 4;
constexpr int kN   = 10000;
constexpr int kT   = 100000;
constexpr int kD   = 128;
constexpr int kH   = 512;
constexpr int kO   = 64;
constexpr int kFIN = 264;   // 2*D + 8
constexpr int kKP  = 288;   // K padded to 9 tiles of 32
constexpr int kRows = kB * kT;  // 400000

// Workspace layout (bytes)
constexpr size_t OFF_W1T = 0;                                // f16 [512][288]
constexpr size_t OFF_W2T = OFF_W1T + (size_t)kH * kKP * 2;   // f16 [64][512]
constexpr size_t OFF_S1  = OFF_W2T + (size_t)kO * kH * 2;
constexpr size_t OFF_T1  = OFF_S1 + (size_t)kH * 4;
constexpr size_t OFF_S2  = OFF_T1 + (size_t)kH * 4;
constexpr size_t OFF_T2  = OFF_S2 + (size_t)kO * 4;

// ---------------- prep kernels ----------------
__global__ void prep_w1(const float* __restrict__ W1, _Float16* __restrict__ W1T) {
  int idx = blockIdx.x * 256 + threadIdx.x;            // over 512*288
  if (idx >= kH * kKP) return;
  int n = idx / kKP, k = idx % kKP;
  float v = (k < kFIN) ? W1[(size_t)k * kH + n] : 0.0f;
  W1T[idx] = (_Float16)v;
}

__global__ void prep_w2(const float* __restrict__ W2, _Float16* __restrict__ W2T) {
  int idx = blockIdx.x * 256 + threadIdx.x;            // over 64*512
  if (idx >= kO * kH) return;
  int o = idx / kH, k = idx % kH;
  W2T[idx] = (_Float16)W2[(size_t)k * kO + o];
}

__global__ void prep_scale(const float* b1, const float* g1, const float* be1,
                           const float* m1, const float* v1,
                           const float* b2, const float* g2, const float* be2,
                           const float* m2, const float* v2,
                           float* s1, float* t1, float* s2, float* t2) {
  int i = blockIdx.x * 256 + threadIdx.x;
  if (i < kH) {
    float s = g1[i] * rsqrtf(v1[i] + 1e-3f);
    s1[i] = s;
    t1[i] = (b1[i] - m1[i]) * s + be1[i];
  } else if (i < kH + kO) {
    int j = i - kH;
    float s = g2[j] * rsqrtf(v2[j] + 1e-3f);
    s2[j] = s;
    t2[j] = (b2[j] - m2[j]) * s + be2[j];
  }
}

// ---------------- fused kernel ----------------
// 2 waves per block, 32 rows (two 16-row tiles) per wave: every B fragment
// (W1T and W2T) is loaded once and feeds two WMMAs.
constexpr int WAVES = 2;
constexpr int XSTR  = kKP * 2;       // 576 B per x row
constexpr int XB    = 16 * XSTR;     // 9216 B per 16-row x tile (reused as y f32)
constexpr int HB    = 16 * 32 * 2;   // 1024 B h staging per tile
constexpr int PW    = 2 * XB + 2 * HB;  // 20480 B per wave

struct __align__(8) H4s { _Float16 a, b, c, d; };

__global__ __launch_bounds__(WAVES * 32) void fused_kernel(
    const float* __restrict__ obj_vecs, const float* __restrict__ box_pred,
    const int* __restrict__ edges,
    const _Float16* __restrict__ W1T, const _Float16* __restrict__ W2T,
    const float* __restrict__ s1, const float* __restrict__ t1,
    const float* __restrict__ s2, const float* __restrict__ t2,
    float* __restrict__ out)
{
  __shared__ __align__(16) char smem[WAVES * PW];   // 40960 B
  const int lane = threadIdx.x & 31;
  const int wave = threadIdx.x >> 5;
  const int tile = blockIdx.x * WAVES + wave;
  const int rowBase = tile * 32;                 // 32 | T, so no batch straddle
  const int b  = rowBase / kT;
  const int t0 = rowBase - b * kT;

  char* base = smem + wave * PW;                 // [x0 | x1 | h0 | h1]
  char* hbase = base + 2 * XB;

  const float* __restrict__ vbase = obj_vecs + (size_t)b * kN * kD;
  const float* __restrict__ bbase = box_pred + (size_t)b * kN * 4;
  const int2*  __restrict__ ebase = (const int2*)edges + (size_t)b * kT + t0;

  // ---- stage x: 2 tiles x 16 rows x 288 f16 (boxes | vecs | zero pad) ----
  for (int r = 0; r < 32; ++r) {
    int2 e = ebase[r];
    _Float16* xr = (_Float16*)(base + (r >> 4) * XB + (r & 15) * XSTR);
    const float4 sv = *(const float4*)(vbase + (size_t)e.x * kD + lane * 4);
    const float4 ov = *(const float4*)(vbase + (size_t)e.y * kD + lane * 4);
    H4s hs = {(_Float16)sv.x, (_Float16)sv.y, (_Float16)sv.z, (_Float16)sv.w};
    H4s ho = {(_Float16)ov.x, (_Float16)ov.y, (_Float16)ov.z, (_Float16)ov.w};
    *(H4s*)(xr + 8   + lane * 4) = hs;           // s_vecs at features [8,136)
    *(H4s*)(xr + 136 + lane * 4) = ho;           // o_vecs at features [136,264)
    if (lane < 4)      xr[lane] = (_Float16)bbase[(size_t)e.x * 4 + lane];
    else if (lane < 8) xr[lane] = (_Float16)bbase[(size_t)e.y * 4 + (lane - 4)];
    if (lane < 12) *(unsigned int*)(xr + kFIN + lane * 2) = 0u;  // pad 264..287
  }

  const int mrow  = lane & 15;
  const int khalf = lane >> 4;

  // ---- cache all A fragments for both tiles (A layout: K runs
  //      [khalf*8,+8) and [16+khalf*8,+8) within each 32-wide k-tile) ----
  v16h A1[2][9];
#pragma unroll
  for (int tt = 0; tt < 2; ++tt)
#pragma unroll
    for (int kt = 0; kt < 9; ++kt) {
      const char* p = base + tt * XB + mrow * XSTR + kt * 64 + khalf * 16;
      A1[tt][kt] = ld_frag(p, p + 32);
    }

  v8f Y[2][4] = {};   // two 16x64 output accumulators

  const char* w1 = (const char*)W1T;
  const char* w2 = (const char*)W2T;

  for (int kt2 = 0; kt2 < 16; ++kt2) {   // GEMM2 k-tiles == pairs of GEMM1 n-tiles
#pragma unroll
    for (int sub = 0; sub < 2; ++sub) {
      const int nt1 = kt2 * 2 + sub;
      const int ng  = nt1 * 16 + mrow;
      v8f acc0 = {}, acc1 = {};
      // B layout: K runs [khalf*8,+8) and [16+khalf*8,+8) within each 32-wide k-tile
      const char* wb = w1 + (size_t)ng * XSTR + khalf * 16;
#pragma unroll
      for (int kt = 0; kt < 9; ++kt) {
        const char* p = wb + kt * 64;
        v16h Bf = ld_frag(p, p + 32);               // shared by both tiles
        acc0 = __builtin_amdgcn_wmma_f32_16x16x32_f16(false, A1[0][kt], false, Bf,
                                                      (short)0, acc0, false, false);
        acc1 = __builtin_amdgcn_wmma_f32_16x16x32_f16(false, A1[1][kt], false, Bf,
                                                      (short)0, acc1, false, false);
      }
      // BN1 + ReLU; transpose C-layout -> row-major f16 via per-wave LDS staging
      const float sc = s1[ng], tb = t1[ng];
      _Float16* h0 = (_Float16*)hbase;
      _Float16* h1 = (_Float16*)(hbase + HB);
#pragma unroll
      for (int r = 0; r < 8; ++r) {
        float hv0 = fmaxf(acc0[r] * sc + tb, 0.0f);
        float hv1 = fmaxf(acc1[r] * sc + tb, 0.0f);
        const int hi = (r + khalf * 8) * 32 + sub * 16 + mrow;
        h0[hi] = (_Float16)hv0;
        h1[hi] = (_Float16)hv1;
      }
    }
    // reload as GEMM2 A fragments (in-wave DScnt ordering; no barrier needed)
    const char* pa0 = hbase + mrow * 64 + khalf * 16;
    const char* pa1 = pa0 + HB;
    v16h A2_0 = ld_frag(pa0, pa0 + 32);
    v16h A2_1 = ld_frag(pa1, pa1 + 32);
#pragma unroll
    for (int nt2 = 0; nt2 < 4; ++nt2) {
      const int og = nt2 * 16 + mrow;
      const char* q = w2 + (size_t)og * (kH * 2) + kt2 * 64 + khalf * 16;
      v16h B2 = ld_frag(q, q + 32);                 // shared by both tiles
      Y[0][nt2] = __builtin_amdgcn_wmma_f32_16x16x32_f16(false, A2_0, false, B2,
                                                         (short)0, Y[0][nt2], false, false);
      Y[1][nt2] = __builtin_amdgcn_wmma_f32_16x16x32_f16(false, A2_1, false, B2,
                                                         (short)0, Y[1][nt2], false, false);
    }
  }

  // ---- epilogue: BN2 + ReLU -> LDS (reuse x buffers as two 16x64 f32) ----
#pragma unroll
  for (int tt = 0; tt < 2; ++tt) {
    float* ybuf = (float*)(base + tt * XB);
#pragma unroll
    for (int nt2 = 0; nt2 < 4; ++nt2) {
      const int og = nt2 * 16 + mrow;
      const float sc = s2[og], tb = t2[og];
#pragma unroll
      for (int r = 0; r < 8; ++r) {
        float yv = fmaxf(Y[tt][nt2][r] * sc + tb, 0.0f);
        ybuf[(r + khalf * 8) * 64 + og] = yv;
      }
    }
  }

  // max(softmax(y)) == 1 / sum(exp(y - ymax)); one row per lane, all 32 lanes:
  // lanes 0-15 -> tile 0 rows 0-15, lanes 16-31 -> tile 1 rows 0-15.
  {
    const float4* yr = (const float4*)((const float*)(base + khalf * XB) + mrow * 64);
    float mx = 0.0f;   // all values >= 0 after ReLU
#pragma unroll
    for (int i = 0; i < 16; ++i) {
      float4 v = yr[i];
      mx = fmaxf(mx, fmaxf(fmaxf(v.x, v.y), fmaxf(v.z, v.w)));
    }
    float sum = 0.0f;
#pragma unroll
    for (int i = 0; i < 16; ++i) {
      float4 v = yr[i];
      sum += __expf(v.x - mx) + __expf(v.y - mx) + __expf(v.z - mx) + __expf(v.w - mx);
    }
    out[rowBase + lane] = 1.0f / sum;   // lane 0-15: rows 0-15; lane 16-31: rows 16-31
  }
}

// ---------------- launcher ----------------
extern "C" void kernel_launch(void* const* d_in, const int* in_sizes, int n_in,
                              void* d_out, int out_size, void* d_ws, size_t ws_size,
                              hipStream_t stream) {
  (void)in_sizes; (void)n_in; (void)out_size; (void)ws_size;
  const float* obj_vecs = (const float*)d_in[0];
  const float* box_pred = (const float*)d_in[1];
  const int*   edges    = (const int*)d_in[2];
  const float* W1  = (const float*)d_in[3];
  const float* b1  = (const float*)d_in[4];
  const float* g1  = (const float*)d_in[5];
  const float* be1 = (const float*)d_in[6];
  const float* m1  = (const float*)d_in[7];
  const float* v1  = (const float*)d_in[8];
  const float* W2  = (const float*)d_in[9];
  const float* b2  = (const float*)d_in[10];
  const float* g2  = (const float*)d_in[11];
  const float* be2 = (const float*)d_in[12];
  const float* m2  = (const float*)d_in[13];
  const float* v2  = (const float*)d_in[14];

  char* ws = (char*)d_ws;
  _Float16* W1T = (_Float16*)(ws + OFF_W1T);
  _Float16* W2T = (_Float16*)(ws + OFF_W2T);
  float* s1 = (float*)(ws + OFF_S1);
  float* t1 = (float*)(ws + OFF_T1);
  float* s2 = (float*)(ws + OFF_S2);
  float* t2 = (float*)(ws + OFF_T2);

  prep_w1<<<(kH * kKP + 255) / 256, 256, 0, stream>>>(W1, W1T);
  prep_w2<<<(kO * kH + 255) / 256, 256, 0, stream>>>(W2, W2T);
  prep_scale<<<3, 256, 0, stream>>>(b1, g1, be1, m1, v1,
                                    b2, g2, be2, m2, v2, s1, t1, s2, t2);

  fused_kernel<<<kRows / (32 * WAVES), WAVES * 32, 0, stream>>>(
      obj_vecs, box_pred, edges, W1T, W2T, s1, t1, s2, t2, (float*)d_out);
}
